// SimplePointTransformer_40613210751458
// MI455X (gfx1250) — matrix operand order repaired and checked by reading, not compiled, not run on hardware
//
#include <hip/hip_runtime.h>
#include <hip/hip_bf16.h>

// ---------------------------------------------------------------------------
// SimplePointTransformer on gfx1250 (MI455X).
// GEMMs use V_WMMA_F32_16X16X32_F16: f32 data converted to f16 fragments
// in-register (v_cvt_pk_f16_f32), f32 accumulate. Weights staged in LDS
// transposed so each B fragment is two ds_load_b128; A fragments are
// four global_load_b128 per k-chunk.
// ---------------------------------------------------------------------------

typedef __attribute__((ext_vector_type(16))) _Float16 v16h;
typedef __attribute__((ext_vector_type(8)))  _Float16 v8h;
typedef __attribute__((ext_vector_type(8)))  float    v8f;

constexpr int NPTS = 16384;   // points
constexpr int SPTS = 4096;    // fps samples
constexpr int CIN  = 32;      // in channels
constexpr int CO   = 64;      // out channels
constexpr int KNB  = 16;      // knn neighbors
constexpr float BN_EPS = 1e-5f;

// ---------------------------------------------------------------------------
// WMMA GEMM: C[rows x 64] = A[rows x KDIM] @ W[KDIM x 64] (+ bias)
// Block = 256 threads = 8 waves; each wave owns one 16-row strip and all four
// 16-col tiles (A fragment reused 4x). Fragment layouts per ISA 7.12.2:
//   A (16x32 f16): lane&15 = M; a[e] is K = kc + half*8 + e (e<8),
//                  K = kc + 16 + half*8 + (e-8) (e>=8)  -> two contig f32 runs
//   B (32x16 f16): lane&15 = N; b[e] has the same K as a[e] -> two contig 8-half runs
//   C (16x16 f32): VGPR j: M = half*8 + j, N = lane&15
// ---------------------------------------------------------------------------
template <int KDIM>
__global__ void k_gemm_wmma(const float* __restrict__ A, const float* __restrict__ W,
                            const float* __restrict__ bias, float* __restrict__ C,
                            int rows) {
  constexpr int NC = 64;
  constexpr int LSTR = KDIM + 8;                 // padded LDS row stride (halfs)
  __shared__ __align__(16) _Float16 Wl[NC * LSTR];

  // stage W (KDIM x 64 row-major f32) transposed into Wl[n][k] as f16
  for (int e = threadIdx.x; e < KDIM * NC; e += blockDim.x) {
    const int k = e >> 6, n = e & 63;
    Wl[n * LSTR + k] = (_Float16)W[e];
  }
  __syncthreads();

  const int lane = threadIdx.x & 31;
  const int wave = threadIdx.x >> 5;
  const int tm = (blockIdx.x * 8 + wave) << 4;
  if (tm >= rows) return;                        // wave-uniform (EXEC all-1s)
  const int m    = lane & 15;
  const int half = lane >> 4;

  v8f acc[4] = {v8f{}, v8f{}, v8f{}, v8f{}};
#pragma unroll
  for (int kc = 0; kc < KDIM; kc += 32) {
    // A fragment: two contiguous 8-float runs -> 4x global_load_b128
    const float* Ar = &A[(size_t)(tm + m) * KDIM + kc + half * 8];
    const float4 c0 = *(const float4*)(Ar);
    const float4 c1 = *(const float4*)(Ar + 4);
    const float4 c2 = *(const float4*)(Ar + 16);
    const float4 c3 = *(const float4*)(Ar + 20);
    v16h a;
    a[0]  = (_Float16)c0.x; a[1]  = (_Float16)c0.y;
    a[2]  = (_Float16)c0.z; a[3]  = (_Float16)c0.w;
    a[4]  = (_Float16)c1.x; a[5]  = (_Float16)c1.y;
    a[6]  = (_Float16)c1.z; a[7]  = (_Float16)c1.w;
    a[8]  = (_Float16)c2.x; a[9]  = (_Float16)c2.y;
    a[10] = (_Float16)c2.z; a[11] = (_Float16)c2.w;
    a[12] = (_Float16)c3.x; a[13] = (_Float16)c3.y;
    a[14] = (_Float16)c3.z; a[15] = (_Float16)c3.w;
#pragma unroll
    for (int nt = 0; nt < 4; ++nt) {
      // B fragment: two 8-half runs (K = half*8.., 16+half*8..) at fixed column -> 2x ds_load_b128
      const _Float16* Bs = &Wl[(nt * 16 + m) * LSTR + kc + half * 8];
      const v8h lo = *(const v8h*)Bs;
      const v8h hi = *(const v8h*)(Bs + 16);
      v16h b;
#pragma unroll
      for (int e = 0; e < 8; ++e) { b[e] = lo[e]; b[8 + e] = hi[e]; }
      acc[nt] = __builtin_amdgcn_wmma_f32_16x16x32_f16(false, a, false, b,
                                                       (short)0, acc[nt],
                                                       false, false);
    }
  }
#pragma unroll
  for (int nt = 0; nt < 4; ++nt) {
    const int col = nt * 16 + m;
    const float bv = bias ? bias[col] : 0.0f;    // hoisted: one load per column
#pragma unroll
    for (int j = 0; j < 8; ++j)
      C[(size_t)(tm + half * 8 + j) * NC + col] = acc[nt][j] + bv;
  }
}

// ---------------------------------------------------------------------------
// KNN: per-thread register top-K over LDS-tiled candidates.
// ---------------------------------------------------------------------------
template <int KK, bool STORE_D>
__global__ void k_knn(const float* __restrict__ q, int nq,
                      const float* __restrict__ data, int nd,
                      int* __restrict__ idxOut, float* __restrict__ distOut) {
  constexpr int TILE = 256;
  __shared__ float sx[TILE], sy[TILE], sz[TILE];
  const int qi = blockIdx.x * blockDim.x + threadIdx.x;
  const bool active = qi < nq;
  float qx = 0.f, qy = 0.f, qz = 0.f;
  if (active) { qx = q[qi * 3]; qy = q[qi * 3 + 1]; qz = q[qi * 3 + 2]; }

  float bestD[KK];
  int   bestI[KK];
#pragma unroll
  for (int j = 0; j < KK; ++j) { bestD[j] = 3.0e38f; bestI[j] = 0; }

  for (int base = 0; base < nd; base += TILE) {
    const int li = base + (int)threadIdx.x;
    if (li < nd) {
      sx[threadIdx.x] = data[li * 3];
      sy[threadIdx.x] = data[li * 3 + 1];
      sz[threadIdx.x] = data[li * 3 + 2];
    }
    if (base + TILE < nd)  // prefetch next tile (global_prefetch_b8)
      __builtin_prefetch(&data[(base + TILE + (int)threadIdx.x) * 3], 0, 0);
    __syncthreads();
    if (active) {
      const int lim = (nd - base < TILE) ? (nd - base) : TILE;
      for (int i = 0; i < lim; ++i) {
        const float dx = qx - sx[i], dy = qy - sy[i], dz = qz - sz[i];
        const float d = dx * dx + dy * dy + dz * dz;
        if (d < bestD[KK - 1]) {
          bestD[KK - 1] = d;
          bestI[KK - 1] = base + i;
#pragma unroll
          for (int j = KK - 1; j > 0; --j) {
            if (bestD[j] < bestD[j - 1]) {
              float td = bestD[j]; bestD[j] = bestD[j - 1]; bestD[j - 1] = td;
              int   ti = bestI[j]; bestI[j] = bestI[j - 1]; bestI[j - 1] = ti;
            }
          }
        }
      }
    }
    __syncthreads();
  }
  if (active) {
#pragma unroll
    for (int j = 0; j < KK; ++j) {
      idxOut[qi * KK + j] = bestI[j];
      if constexpr (STORE_D) distOut[qi * KK + j] = bestD[j];
    }
  }
}

// ---------------------------------------------------------------------------
// Farthest point sampling: one persistent 1024-thread block, points + dmin in
// registers (16 per thread), LDS tree argmax per step. 4096 serial steps.
// ---------------------------------------------------------------------------
__global__ void k_fps(const float* __restrict__ points, int* __restrict__ s_idx) {
  constexpr int T = 1024, PP = NPTS / 1024;
  __shared__ float red_d[T];
  __shared__ int   red_i[T];
  __shared__ float farPt[3];
  const int t = threadIdx.x;
  float px[PP], py[PP], pz[PP], dmin[PP];
#pragma unroll
  for (int p = 0; p < PP; ++p) {
    const int i = p * T + t;
    px[p] = points[i * 3]; py[p] = points[i * 3 + 1]; pz[p] = points[i * 3 + 2];
    dmin[p] = 1e10f;
  }
  if (t == 0) {
    s_idx[0] = 0;
    farPt[0] = points[0]; farPt[1] = points[1]; farPt[2] = points[2];
  }
  __syncthreads();
  for (int it = 1; it < SPTS; ++it) {
    const float fx = farPt[0], fy = farPt[1], fz = farPt[2];
    float ld = -1.f; int li = 0;
#pragma unroll
    for (int p = 0; p < PP; ++p) {
      const float dx = px[p] - fx, dy = py[p] - fy, dz = pz[p] - fz;
      const float d = dx * dx + dy * dy + dz * dz;
      dmin[p] = fminf(dmin[p], d);
      if (dmin[p] > ld) { ld = dmin[p]; li = p * T + t; }
    }
    red_d[t] = ld; red_i[t] = li;
    __syncthreads();
    for (int off = T >> 1; off > 0; off >>= 1) {
      if (t < off && red_d[t + off] > red_d[t]) {
        red_d[t] = red_d[t + off]; red_i[t] = red_i[t + off];
      }
      __syncthreads();
    }
    if (t == 0) {
      const int far = red_i[0];
      s_idx[it] = far;
      farPt[0] = points[far * 3]; farPt[1] = points[far * 3 + 1];
      farPt[2] = points[far * 3 + 2];
    }
    __syncthreads();
  }
}

// ---------------------------------------------------------------------------
// BatchNorm stats over rows x 64 (LDS partials -> global float atomics).
// stats[0:64]=sum, stats[64:128]=sumsq (pre-zeroed via hipMemsetAsync).
// ---------------------------------------------------------------------------
__global__ void k_bn_stats64(const float* __restrict__ X, int rows,
                             float* __restrict__ stats) {
  __shared__ float ssum[CO], ssq[CO];
  const int t = threadIdx.x;
  if (t < CO) { ssum[t] = 0.f; ssq[t] = 0.f; }
  __syncthreads();
  const int c = t & 63;
  const int rstart = blockIdx.x * (blockDim.x >> 6) + (t >> 6);
  const int rstep = gridDim.x * (blockDim.x >> 6);
  float s = 0.f, q = 0.f;
  for (int r = rstart; r < rows; r += rstep) {
    const float x = X[(size_t)r * CO + c];
    s += x; q += x * x;
  }
  atomicAdd(&ssum[c], s);
  atomicAdd(&ssq[c], q);
  __syncthreads();
  if (t < CO) { atomicAdd(&stats[t], ssum[t]); atomicAdd(&stats[CO + t], ssq[t]); }
}

__global__ void k_bn_relu64(const float* __restrict__ X, float* __restrict__ Y,
                            int rows, const float* __restrict__ stats,
                            const float* __restrict__ g, const float* __restrict__ b) {
  const size_t i = (size_t)blockIdx.x * blockDim.x + threadIdx.x;
  const size_t total = (size_t)rows * CO;
  if (i >= total) return;
  const int c = (int)(i & 63);
  const float inv = 1.0f / (float)rows;
  const float m = stats[c] * inv;
  const float var = stats[CO + c] * inv - m * m;
  const float y = (X[i] - m) * rsqrtf(var + BN_EPS) * g[c] + b[c];
  Y[i] = fmaxf(y, 0.f);
}

// ---------------------------------------------------------------------------
// rel = p[n] - p[knn[n,k]]; t = rel @ Wp1 (3x3); accumulate 3-ch BN stats.
// ---------------------------------------------------------------------------
__global__ void k_rel_stats(const float* __restrict__ points,
                            const int* __restrict__ knn_idx,
                            const float* __restrict__ Wp1,
                            float* __restrict__ t_rel, float* __restrict__ stats6) {
  __shared__ float s6[6];
  if (threadIdx.x < 6) s6[threadIdx.x] = 0.f;
  __syncthreads();
  const int i = blockIdx.x * blockDim.x + threadIdx.x;  // over NPTS*KNB
  float ls[6] = {0.f, 0.f, 0.f, 0.f, 0.f, 0.f};
  if (i < NPTS * KNB) {
    const int n = i >> 4;
    const int j = knn_idx[i];
    const float rx = points[n * 3] - points[j * 3];
    const float ry = points[n * 3 + 1] - points[j * 3 + 1];
    const float rz = points[n * 3 + 2] - points[j * 3 + 2];
#pragma unroll
    for (int c = 0; c < 3; ++c) {
      const float tc = rx * Wp1[c] + ry * Wp1[3 + c] + rz * Wp1[6 + c];
      t_rel[i * 3 + c] = tc;
      ls[c] = tc; ls[3 + c] = tc * tc;
    }
  }
#pragma unroll
  for (int c = 0; c < 6; ++c) atomicAdd(&s6[c], ls[c]);
  __syncthreads();
  if (threadIdx.x < 6) atomicAdd(&stats6[threadIdx.x], s6[threadIdx.x]);
}

// rpe[n,k,:] = relu(bn3(t_rel)) @ Wp2 (3x64)
__global__ void k_rpe(const float* __restrict__ t_rel, const float* __restrict__ stats6,
                      const float* __restrict__ gp, const float* __restrict__ bp,
                      const float* __restrict__ Wp2, float* __restrict__ rpe) {
  const int i = blockIdx.x * blockDim.x + threadIdx.x;  // over NPTS*KNB
  if (i >= NPTS * KNB) return;
  const float invR = 1.0f / (float)(NPTS * KNB);
  float h[3];
#pragma unroll
  for (int c = 0; c < 3; ++c) {
    const float m = stats6[c] * invR;
    const float v = stats6[3 + c] * invR - m * m;
    h[c] = fmaxf((t_rel[i * 3 + c] - m) * rsqrtf(v + BN_EPS) * gp[c] + bp[c], 0.f);
  }
  const size_t base = (size_t)i * CO;
  for (int c = 0; c < CO; ++c)
    rpe[base + c] = h[0] * Wp2[c] + h[1] * Wp2[CO + c] + h[2] * Wp2[2 * CO + c];
}

// x[n,k,c] = f_q[n,c] - f_k[knn[n,k],c] + rpe[n,k,c]
__global__ void k_vecsim(const float* __restrict__ fq, const float* __restrict__ fk,
                         const float* __restrict__ rpe, const int* __restrict__ knn_idx,
                         float* __restrict__ out) {
  const size_t i = (size_t)blockIdx.x * blockDim.x + threadIdx.x;
  if (i >= (size_t)NPTS * KNB * CO) return;
  const int c = (int)(i & 63);
  const size_t nk = i >> 6;
  const int n = (int)(nk >> 4);
  const int j = knn_idx[nk];
  out[i] = fq[(size_t)n * CO + c] - fk[(size_t)j * CO + c] + rpe[i];
}

// softmax over K per (n,c); skip[n,c] = sum_k w * (f_v[knn] + rpe)
__global__ void k_attn_out(const float* __restrict__ logits, const float* __restrict__ fv,
                           const float* __restrict__ rpe, const int* __restrict__ knn_idx,
                           float* __restrict__ skip) {
  const int i = blockIdx.x * blockDim.x + threadIdx.x;  // over NPTS*CO
  if (i >= NPTS * CO) return;
  const int c = i & 63;
  const int n = i >> 6;
  float mx = -3.0e38f;
#pragma unroll
  for (int k = 0; k < KNB; ++k)
    mx = fmaxf(mx, logits[(size_t)(n * KNB + k) * CO + c]);
  float e[KNB], s = 0.f;
#pragma unroll
  for (int k = 0; k < KNB; ++k) {
    e[k] = __expf(logits[(size_t)(n * KNB + k) * CO + c] - mx);
    s += e[k];
  }
  const float inv = 1.0f / s;
  float acc = 0.f;
#pragma unroll
  for (int k = 0; k < KNB; ++k) {
    const int j = knn_idx[n * KNB + k];
    acc += e[k] * inv * (fv[(size_t)j * CO + c] + rpe[(size_t)(n * KNB + k) * CO + c]);
  }
  skip[i] = acc;
}

// gathers / pooling / final
__global__ void k_gather64(const float* __restrict__ src, const int* __restrict__ idx,
                           float* __restrict__ out, int rows) {
  const size_t i = (size_t)blockIdx.x * blockDim.x + threadIdx.x;
  if (i >= (size_t)rows * CO) return;
  const int c = (int)(i & 63);
  const int r = (int)(i >> 6);
  out[i] = src[(size_t)idx[r] * CO + c];
}

__global__ void k_gather_pts(const float* __restrict__ points, const int* __restrict__ idx,
                             float* __restrict__ out) {
  const int i = blockIdx.x * blockDim.x + threadIdx.x;  // over SPTS*3
  if (i >= SPTS * 3) return;
  out[i] = points[idx[i / 3] * 3 + (i % 3)];
}

__global__ void k_maxpool(const float* __restrict__ X, float* __restrict__ out) {
  const int i = blockIdx.x * blockDim.x + threadIdx.x;  // over SPTS*CO
  if (i >= SPTS * CO) return;
  const int c = i & 63;
  const int s = i >> 6;
  float m = -3.0e38f;
#pragma unroll
  for (int k = 0; k < KNB; ++k)
    m = fmaxf(m, X[(size_t)(s * KNB + k) * CO + c]);
  out[i] = m;
}

// out = interp(df via 3-NN inverse-distance) + (skip @ Wup + bup)  (U precomputed)
__global__ void k_final(const float* __restrict__ df, const int* __restrict__ idx3,
                        const float* __restrict__ d3, const float* __restrict__ U,
                        float* __restrict__ out) {
  const int i = blockIdx.x * blockDim.x + threadIdx.x;  // over NPTS*CO
  if (i >= NPTS * CO) return;
  const int c = i & 63;
  const int n = i >> 6;
  const float r0 = 1.0f / (d3[n * 3] + 1e-8f);
  const float r1 = 1.0f / (d3[n * 3 + 1] + 1e-8f);
  const float r2 = 1.0f / (d3[n * 3 + 2] + 1e-8f);
  const float inv = 1.0f / (r0 + r1 + r2);
  const float interp = (r0 * df[(size_t)idx3[n * 3] * CO + c] +
                        r1 * df[(size_t)idx3[n * 3 + 1] * CO + c] +
                        r2 * df[(size_t)idx3[n * 3 + 2] * CO + c]) * inv;
  out[i] = interp + U[i];
}

// ---------------------------------------------------------------------------
// Host side
// ---------------------------------------------------------------------------
static inline void gemm(const float* A, const float* W, const float* bias, float* C,
                        int rows, int kdim, hipStream_t s) {
  const int blocks = rows / 128;  // 8 row-tiles (waves) per 256-thread block
  if (kdim == 32) k_gemm_wmma<32><<<blocks, 256, 0, s>>>(A, W, bias, C, rows);
  else            k_gemm_wmma<64><<<blocks, 256, 0, s>>>(A, W, bias, C, rows);
}

extern "C" void kernel_launch(void* const* d_in, const int* in_sizes, int n_in,
                              void* d_out, int out_size, void* d_ws, size_t ws_size,
                              hipStream_t stream) {
  const float* points = (const float*)d_in[0];
  const float* feats  = (const float*)d_in[1];
  const float* Wq  = (const float*)d_in[2];
  const float* Wk  = (const float*)d_in[3];
  const float* Wv  = (const float*)d_in[4];
  const float* ga1 = (const float*)d_in[5];
  const float* ba1 = (const float*)d_in[6];
  const float* Wa1 = (const float*)d_in[7];
  const float* ga2 = (const float*)d_in[8];
  const float* ba2 = (const float*)d_in[9];
  const float* Wa2 = (const float*)d_in[10];
  const float* b_a2 = (const float*)d_in[11];
  const float* Wp1 = (const float*)d_in[12];
  const float* gp  = (const float*)d_in[13];
  const float* bp  = (const float*)d_in[14];
  const float* Wp2 = (const float*)d_in[15];
  const float* Wd1 = (const float*)d_in[16];
  const float* gd1 = (const float*)d_in[17];
  const float* bd1 = (const float*)d_in[18];
  const float* Wd2 = (const float*)d_in[19];
  const float* gd2 = (const float*)d_in[20];
  const float* bd2 = (const float*)d_in[21];
  const float* Wup = (const float*)d_in[22];
  const float* bup = (const float*)d_in[23];
  const float* Wdn = (const float*)d_in[24];
  const float* bdn = (const float*)d_in[25];
  float* out = (float*)d_out;

  // workspace layout (bump allocation; every offset is a multiple of 16 bytes)
  float* fq      = (float*)d_ws;                        // NPTS*CO
  float* fk      = fq + (size_t)NPTS * CO;              // NPTS*CO
  float* fv      = fk + (size_t)NPTS * CO;              // NPTS*CO
  int*   knn_idx = (int*)(fv + (size_t)NPTS * CO);      // NPTS*KNB
  float* t_rel   = (float*)(knn_idx + NPTS * KNB);      // NPTS*KNB*3
  float* rpe     = t_rel + (size_t)NPTS * KNB * 3;      // NPTS*KNB*CO
  float* bufA    = rpe + (size_t)NPTS * KNB * CO;       // NPTS*KNB*CO
  float* bufB    = bufA + (size_t)NPTS * KNB * CO;      // NPTS*KNB*CO
  float* skip    = bufB + (size_t)NPTS * KNB * CO;      // NPTS*CO
  int*   s_idx   = (int*)(skip + (size_t)NPTS * CO);    // SPTS
  float* s_pts   = (float*)(s_idx + SPTS);              // SPTS*3
  int*   idx2    = (int*)(s_pts + SPTS * 3);            // SPTS*KNB
  float* down_f  = (float*)(idx2 + SPTS * KNB);         // SPTS*CO
  float* df      = down_f + (size_t)SPTS * CO;          // SPTS*CO
  int*   idx3    = (int*)(df + (size_t)SPTS * CO);      // NPTS*3
  float* d3      = (float*)(idx3 + NPTS * 3);           // NPTS*3
  float* U       = d3 + (size_t)NPTS * 3;               // NPTS*CO
  float* stats   = U + (size_t)NPTS * CO;               // 128
  float* stats6  = stats + 128;                         // 8

  const int NK = NPTS * KNB;

  // ---- PointTransformerLayer ----
  gemm(feats, Wq, nullptr, fq, NPTS, CIN, stream);
  gemm(feats, Wk, nullptr, fk, NPTS, CIN, stream);
  gemm(feats, Wv, nullptr, fv, NPTS, CIN, stream);

  k_knn<KNB, false><<<NPTS / 256, 256, 0, stream>>>(points, NPTS, points, NPTS,
                                                    knn_idx, nullptr);

  hipMemsetAsync(stats6, 0, 8 * sizeof(float), stream);
  k_rel_stats<<<NK / 256, 256, 0, stream>>>(points, knn_idx, Wp1, t_rel, stats6);
  k_rpe<<<NK / 256, 256, 0, stream>>>(t_rel, stats6, gp, bp, Wp2, rpe);

  k_vecsim<<<(int)(((size_t)NK * CO) / 256), 256, 0, stream>>>(fq, fk, rpe, knn_idx,
                                                               bufA);
  // bn1 + relu -> bufB ; @Wa1 -> bufA
  hipMemsetAsync(stats, 0, 128 * sizeof(float), stream);
  k_bn_stats64<<<256, 256, 0, stream>>>(bufA, NK, stats);
  k_bn_relu64<<<(int)(((size_t)NK * CO) / 256), 256, 0, stream>>>(bufA, bufB, NK,
                                                                  stats, ga1, ba1);
  gemm(bufB, Wa1, nullptr, bufA, NK, CO, stream);
  // bn2 + relu -> bufB ; @Wa2 + b_a2 -> bufA (attention logits)
  hipMemsetAsync(stats, 0, 128 * sizeof(float), stream);
  k_bn_stats64<<<256, 256, 0, stream>>>(bufA, NK, stats);
  k_bn_relu64<<<(int)(((size_t)NK * CO) / 256), 256, 0, stream>>>(bufA, bufB, NK,
                                                                  stats, ga2, ba2);
  gemm(bufB, Wa2, b_a2, bufA, NK, CO, stream);

  k_attn_out<<<NPTS * CO / 256, 256, 0, stream>>>(bufA, fv, rpe, knn_idx, skip);

  // ---- TransitionDown ----
  k_fps<<<1, 1024, 0, stream>>>(points, s_idx);
  k_gather_pts<<<(SPTS * 3 + 255) / 256, 256, 0, stream>>>(points, s_idx, s_pts);
  k_knn<KNB, false><<<SPTS / 256, 256, 0, stream>>>(s_pts, SPTS, points, NPTS,
                                                    idx2, nullptr);
  const int SK = SPTS * KNB;
  k_gather64<<<SK * CO / 256, 256, 0, stream>>>(skip, idx2, bufA, SK);
  gemm(bufA, Wd1, nullptr, bufB, SK, CO, stream);
  hipMemsetAsync(stats, 0, 128 * sizeof(float), stream);
  k_bn_stats64<<<256, 256, 0, stream>>>(bufB, SK, stats);
  k_bn_relu64<<<SK * CO / 256, 256, 0, stream>>>(bufB, bufA, SK, stats, gd1, bd1);
  gemm(bufA, Wd2, nullptr, bufB, SK, CO, stream);
  hipMemsetAsync(stats, 0, 128 * sizeof(float), stream);
  k_bn_stats64<<<256, 256, 0, stream>>>(bufB, SK, stats);
  k_bn_relu64<<<SK * CO / 256, 256, 0, stream>>>(bufB, bufA, SK, stats, gd2, bd2);
  k_maxpool<<<SPTS * CO / 256, 256, 0, stream>>>(bufA, down_f);

  // ---- TransitionUp ----
  gemm(down_f, Wdn, bdn, df, SPTS, CO, stream);
  k_knn<3, true><<<NPTS / 256, 256, 0, stream>>>(points, NPTS, s_pts, SPTS, idx3, d3);
  gemm(skip, Wup, bup, U, NPTS, CO, stream);
  k_final<<<NPTS * CO / 256, 256, 0, stream>>>(df, idx3, d3, U, out);

  (void)in_sizes; (void)n_in; (void)out_size; (void)ws_size;
}
